// NonFirstLayerAggregator_55362128445577
// MI455X (gfx1250) — matrix operand order repaired and checked by reading, not compiled, not run on hardware
//
#include <hip/hip_runtime.h>

#define NN    100000
#define UU    200000
#define DEG   16
#define DD    64
#define HH    4
#define LRA   0.2f

typedef __attribute__((ext_vector_type(16))) _Float16 v16h;
typedef __attribute__((ext_vector_type(8)))  float    v8f;
typedef __attribute__((ext_vector_type(2)))  _Float16 f16x2;

__device__ __forceinline__ float wave_max32(float v) {
#pragma unroll
    for (int off = 16; off > 0; off >>= 1)
        v = fmaxf(v, __shfl_xor(v, off, 32));
    return v;
}
__device__ __forceinline__ float wave_sum32(float v) {
#pragma unroll
    for (int off = 16; off > 0; off >>= 1)
        v += __shfl_xor(v, off, 32);
    return v;
}
// Uniform-lane broadcast into SGPRs (lane index is a literal in unrolled loops).
__device__ __forceinline__ float bcast_f(float v, int l) {
    return __int_as_float(__builtin_amdgcn_readlane(__float_as_int(v), l));
}
__device__ __forceinline__ int bcast_i(int v, int l) {
    return __builtin_amdgcn_readlane(v, l);
}

// ---------------------------------------------------------------------------
// Kernel 1: h[head][u][d] = sum_k x[u][k] * W[head][k][d], stored as f16.
// One wave computes a 16(M) x 64(N) tile for one head using
// v_wmma_f32_16x16x32_f16 (2 K-chunks x 4 N-tiles = 8 WMMA / wave).
// ---------------------------------------------------------------------------
__global__ __launch_bounds__(256) void gemm_h_kernel(
    const float* __restrict__ x, const float* __restrict__ W,
    _Float16* __restrict__ h)
{
    const int tiles = UU / 16;                       // 12500
    int w = blockIdx.x * 8 + (threadIdx.x >> 5);     // wave job id
    if (w >= tiles * HH) return;
    const int head = w / tiles;
    const int tile = w % tiles;
    const int lane = threadIdx.x & 31;
    const int l15  = lane & 15;
    const int hi   = (lane & 16) ? 1 : 0;

    const float* __restrict__ Wh = W + (size_t)head * DD * DD;
    const int mrow = tile * 16 + l15;

    v8f acc[4];
#pragma unroll
    for (int nt = 0; nt < 4; ++nt) acc[nt] = (v8f)(0.0f);

#pragma unroll
    for (int kc = 0; kc < DD; kc += 32) {
        // ---- A tile (16x32 f16): lane<16 holds row M=l15, K = kc+0..15
        //                          lane>=16 holds row M=l15, K = kc+16..31
        const int kbA = kc + (hi ? 16 : 0);
        const float* __restrict__ xp = x + (size_t)mrow * DD + kbA;
        float4 a0 = *(const float4*)(xp);
        float4 a1 = *(const float4*)(xp + 4);
        float4 a2 = *(const float4*)(xp + 8);
        float4 a3 = *(const float4*)(xp + 12);
        v16h A;
        A[0]  = (_Float16)a0.x; A[1]  = (_Float16)a0.y;
        A[2]  = (_Float16)a0.z; A[3]  = (_Float16)a0.w;
        A[4]  = (_Float16)a1.x; A[5]  = (_Float16)a1.y;
        A[6]  = (_Float16)a1.z; A[7]  = (_Float16)a1.w;
        A[8]  = (_Float16)a2.x; A[9]  = (_Float16)a2.y;
        A[10] = (_Float16)a2.z; A[11] = (_Float16)a2.w;
        A[12] = (_Float16)a3.x; A[13] = (_Float16)a3.y;
        A[14] = (_Float16)a3.z; A[15] = (_Float16)a3.w;

#pragma unroll
        for (int nt = 0; nt < 4; ++nt) {
            // ---- B tile (32x16 f16): lane<16 holds col N=l15, K = kc+0..15
            //                          lane>=16 holds col N=l15, K = kc+16..31
            const int ncol = nt * 16 + l15;
            const int kbB  = kc + (hi ? 16 : 0);
            const float* __restrict__ wp = Wh + (size_t)kbB * DD + ncol;
            v16h B;
#pragma unroll
            for (int t = 0; t < 16; ++t)
                B[t] = (_Float16)wp[(size_t)t * DD];

            acc[nt] = __builtin_amdgcn_wmma_f32_16x16x32_f16(
                false, A, false, B, (short)0, acc[nt], false, false);
        }
    }

    // ---- C store: VGPR v holds (M = v + 8*hi, N = l15) for this lane
    const int mbase = tile * 16 + (hi ? 8 : 0);
#pragma unroll
    for (int nt = 0; nt < 4; ++nt) {
#pragma unroll
        for (int v = 0; v < 8; ++v) {
            h[((size_t)head * UU + mbase + v) * DD + nt * 16 + l15] =
                (_Float16)acc[nt][v];
        }
    }
}

// ---------------------------------------------------------------------------
// Kernel 2: per (u, head) attention scores
//   s[head][u] = { h[u].a_pos[:64], h[u].a_pos[64:], h[u].a_neg[:64], h[u].a_neg[64:] }
// One wave per (u, head); lane owns d = 2*lane, 2*lane+1.
// ---------------------------------------------------------------------------
__global__ __launch_bounds__(256) void scores_kernel(
    const _Float16* __restrict__ h, const float* __restrict__ a_pos,
    const float* __restrict__ a_neg, float* __restrict__ s)
{
    int w = blockIdx.x * 8 + (threadIdx.x >> 5);
    if (w >= UU * HH) return;
    const int head = w / UU;
    const int u    = w % UU;
    const int lane = threadIdx.x & 31;
    const int d    = 2 * lane;

    f16x2 hv = *(const f16x2*)(h + ((size_t)head * UU + u) * DD + d);
    const float h0 = (float)hv.x, h1 = (float)hv.y;
    const float* __restrict__ ap = a_pos + head * 2 * DD;
    const float* __restrict__ an = a_neg + head * 2 * DD;

    float p0 = h0 * ap[d]      + h1 * ap[d + 1];
    float p1 = h0 * ap[DD + d] + h1 * ap[DD + d + 1];
    float p2 = h0 * an[d]      + h1 * an[d + 1];
    float p3 = h0 * an[DD + d] + h1 * an[DD + d + 1];

    p0 = wave_sum32(p0); p1 = wave_sum32(p1);
    p2 = wave_sum32(p2); p3 = wave_sum32(p3);

    if (lane == 0) {
        float4* sp = (float4*)(s + ((size_t)head * UU + u) * 4);
        *sp = make_float4(p0, p1, p2, p3);
    }
}

// ---------------------------------------------------------------------------
// Kernel 3: per-node softmax aggregation over pos(17) + neg(16) edges, all
// 4 heads, mean + ReLU. One wave per node; lane owns d = 2*lane, 2*lane+1.
// Broadcasts att/col to SGPRs via v_readlane so each gather becomes a
// scalar-base (saddr) global_load_b32.
// ---------------------------------------------------------------------------
__global__ __launch_bounds__(256) void aggregate_kernel(
    const int* __restrict__ adj_pos, const int* __restrict__ adj_neg,
    const _Float16* __restrict__ h, const float* __restrict__ s,
    float* __restrict__ out)
{
    int i = blockIdx.x * 8 + (threadIdx.x >> 5);
    if (i >= NN) return;
    const int lane = threadIdx.x & 31;

    const int* __restrict__ colp = adj_pos + (size_t)NN * (DEG + 1); // col half
    const int* __restrict__ coln = adj_neg + (size_t)NN * DEG;       // col half

    // per-lane edge columns (valid index even for inactive lanes)
    const bool pact = (lane <= 16);
    const bool nact = (lane < 16);
    const int cp = (lane < 16) ? colp[(size_t)i * DEG + lane] : i;   // lane16: self
    const int cn = nact ? coln[(size_t)i * DEG + lane] : 0;

    float acc0 = 0.0f, acc1 = 0.0f;

#pragma unroll
    for (int head = 0; head < HH; ++head) {
        const float* __restrict__ sh = s + (size_t)head * UU * 4;
        const _Float16* __restrict__ hh = h + (size_t)head * UU * DD;
        const float s1p = sh[(size_t)i * 4 + 0];
        const float s1n = sh[(size_t)i * 4 + 2];

        // ---- positive edges (16 neighbors + self) ----
        {
            float e = -3.0e38f;
            if (pact) {
                float v = s1p + sh[(size_t)cp * 4 + 1];
                e = (v >= 0.0f) ? v : LRA * v;
            }
            const float m   = wave_max32(e);
            const float ex  = pact ? __expf(e - m) : 0.0f;
            const float den = wave_sum32(ex);
            const float att = ex / den;
#pragma unroll
            for (int j = 0; j < DEG + 1; ++j) {
                const float a = bcast_f(att, j);      // SGPR broadcast
                const int   c = bcast_i(cp, j);       // SGPR broadcast
                f16x2 hv = *(const f16x2*)(hh + (size_t)c * DD + 2 * lane);
                acc0 += a * (float)hv.x;
                acc1 += a * (float)hv.y;
            }
        }
        // ---- negative edges (16 neighbors) ----
        {
            float e = -3.0e38f;
            if (nact) {
                float v = s1n + sh[(size_t)cn * 4 + 3];
                e = (v >= 0.0f) ? v : LRA * v;
            }
            const float m   = wave_max32(e);
            const float ex  = nact ? __expf(e - m) : 0.0f;
            const float den = wave_sum32(ex);
            const float att = ex / den;
#pragma unroll
            for (int j = 0; j < DEG; ++j) {
                const float a = bcast_f(att, j);
                const int   c = bcast_i(cn, j);
                f16x2 hv = *(const f16x2*)(hh + (size_t)c * DD + 2 * lane);
                acc0 += a * (float)hv.x;
                acc1 += a * (float)hv.y;
            }
        }
    }

    float r0 = acc0 * (1.0f / HH);
    float r1 = acc1 * (1.0f / HH);
    r0 = (r0 > 0.0f) ? r0 : 0.0f;
    r1 = (r1 > 0.0f) ? r1 : 0.0f;
    *(float2*)(out + (size_t)i * DD + 2 * lane) = make_float2(r0, r1);
}

// ---------------------------------------------------------------------------
// Kernel 4: pass-through copy of edge_embedding (second tuple output) via the
// CDNA5 async LDS path: global -> LDS -> global, tracked by ASYNCcnt, no
// VGPR data movement. 16 bytes per lane per block-slot; grid covers exactly
// UU*DD floats (12500 blocks x 256 lanes x 16B).
// ---------------------------------------------------------------------------
__global__ __launch_bounds__(256) void copy_ee_kernel(
    const float* __restrict__ src, float* __restrict__ dst)
{
    __shared__ __align__(16) float buf[256 * 4];
    const int tid = threadIdx.x;
    const size_t idx = ((size_t)blockIdx.x * 256 + tid) * 4;  // float index
    const float* gsrc = src + idx;
    float*       gdst = dst + idx;
    // Generic LDS pointer keeps the byte offset in addr[31:0].
    unsigned ldsoff = (unsigned)(uintptr_t)(&buf[tid * 4]);

    asm volatile("global_load_async_to_lds_b128 %0, %1, off"
                 :: "v"(ldsoff), "v"(gsrc) : "memory");
    asm volatile("s_wait_asynccnt 0" ::: "memory");
    asm volatile("global_store_async_from_lds_b128 %0, %1, off"
                 :: "v"(gdst), "v"(ldsoff) : "memory");
    asm volatile("s_wait_asynccnt 0" ::: "memory");
}

// ---------------------------------------------------------------------------
extern "C" void kernel_launch(void* const* d_in, const int* in_sizes, int n_in,
                              void* d_out, int out_size, void* d_ws, size_t ws_size,
                              hipStream_t stream) {
    const float* x    = (const float*)d_in[0];
    const float* ee   = (const float*)d_in[1];
    const int*   adjp = (const int*)d_in[2];
    const int*   adjn = (const int*)d_in[3];
    const float* W    = (const float*)d_in[4];
    const float* ap   = (const float*)d_in[5];
    const float* an   = (const float*)d_in[6];
    float* out = (float*)d_out;

    // workspace layout: f16 h [H][U][D] then f32 scores [H][U][4]
    _Float16* h = (_Float16*)d_ws;
    float*    s = (float*)((char*)d_ws + (size_t)HH * UU * DD * sizeof(_Float16));

    // gemm: 12500 tiles * 4 heads = 50000 wave-jobs, 8 waves/block
    gemm_h_kernel<<<(UU / 16) * HH / 8, 256, 0, stream>>>(x, W, h);
    // scores: 200000 * 4 wave-jobs
    scores_kernel<<<UU * HH / 8, 256, 0, stream>>>(h, ap, an, s);
    // aggregate: 100000 wave-jobs
    aggregate_kernel<<<NN / 8, 256, 0, stream>>>(adjp, adjn, h, s, out);
    // edge_embedding passthrough via async LDS bounce
    copy_ee_kernel<<<UU * DD / (256 * 4), 256, 0, stream>>>(
        ee, out + (size_t)NN * DD);
}
